// DualAdapt_64149631533758
// MI455X (gfx1250) — matrix operand order repaired and checked by reading, not compiled, not run on hardware
//
#include <hip/hip_runtime.h>
#include <hip/hip_bf16.h>

typedef __attribute__((ext_vector_type(16))) _Float16 v16h;
typedef __attribute__((ext_vector_type(8)))  _Float16 v8h;
typedef __attribute__((ext_vector_type(8)))  float    v8f;

#define D_DIM 768
#define DW_PER_ROW (D_DIM / 2)     // 384 dwords of f16 per row
#define A_STRIDE 776               // padded LDS row stride in halves (388 dwords, bank-conflict free)

// ---------------------------------------------------------------------------
// Stage 1: row-wise L2 normalize fp32 -> fp16. One wave32 per row.
// ---------------------------------------------------------------------------
__global__ __launch_bounds__(256)
void rownorm_f16(const float* __restrict__ x, _Float16* __restrict__ out, int rows) {
    int row  = blockIdx.x * 8 + (threadIdx.x >> 5);
    int lane = threadIdx.x & 31;
    if (row >= rows) return;

    const float* xr = x + (size_t)row * D_DIM;
    float v[24];
    float s = 0.0f;
#pragma unroll
    for (int t = 0; t < 24; ++t) {
        v[t] = xr[lane + 32 * t];
        s += v[t] * v[t];
    }
#pragma unroll
    for (int off = 16; off > 0; off >>= 1)
        s += __shfl_xor(s, off);

    float scale = 1.0f / fmaxf(sqrtf(s), 1e-12f);

    _Float16* outr = out + (size_t)row * D_DIM;
#pragma unroll
    for (int t = 0; t < 24; ++t)
        outr[lane + 32 * t] = (_Float16)(v[t] * scale);
}

// ---------------------------------------------------------------------------
// Stage 2: fused cos-sim GEMM + row argmax.
// Block = 256 threads (8 waves) handles 16 query rows vs all P prototype cols.
// Each wave computes TWO 16-col tiles per K pass (shared A load), software-
// pipelined so global/LDS loads overlap the WMMAs.
// ---------------------------------------------------------------------------
__global__ __launch_bounds__(256)
void argmax_gemm(const _Float16* __restrict__ qh,
                 const _Float16* __restrict__ nkh,
                 int* __restrict__ kidx, int P) {
    __shared__ _Float16 As[16 * A_STRIDE];       // ~24.8 KB, padded rows
    __shared__ float red_val[8][16];
    __shared__ int   red_idx[8][16];

    const int mbase = blockIdx.x * 16;
    const int tid   = threadIdx.x;

    // Stage A tile into LDS (coalesced dword copies, padded destination rows).
    {
        const unsigned int* src = (const unsigned int*)(qh + (size_t)mbase * D_DIM);
        unsigned int*       dst = (unsigned int*)As;
        for (int t = tid; t < 16 * DW_PER_ROW; t += 256) {
            int row = t / DW_PER_ROW, dw = t - row * DW_PER_ROW;
            dst[row * (A_STRIDE / 2) + dw] = src[t];
        }
    }
    __syncthreads();

    const int wv   = tid >> 5;
    const int lane = tid & 31;
    const int m    = lane & 15;   // A row within tile / B column within tile
    const int hi   = lane >> 4;   // half-wave selector for K striping
    const int arow = m * A_STRIDE;
    const int hi8  = hi * 8;

    // A lane layout (16-bit A 16x32): lane<16 -> K {k0..k0+15}
    //                                 lane>=16 -> K {k0+16..k0+31}
#define LOAD_A(K0) __builtin_shufflevector(                                   \
        *(const v8h*)&As[arow + (K0) + 2 * hi8],                              \
        *(const v8h*)&As[arow + (K0) + 8 + 2 * hi8],                          \
        0, 1, 2, 3, 4, 5, 6, 7, 8, 9, 10, 11, 12, 13, 14, 15)

    float best[8];
    int   bidx[8];
#pragma unroll
    for (int r = 0; r < 8; ++r) { best[r] = -3.4e38f; bidx[r] = 0; }

    const int ntiles = P / 256;   // dual-tile passes per wave
    for (int jt = 0; jt < ntiles; ++jt) {
        const int cb = wv * 32 + jt * 256;
        // B lane layout (16-bit B 32x16): N = lane%16, K = hi*16..hi*16+15 contiguous
        const _Float16* b0p = nkh + (size_t)(cb + m) * D_DIM + hi * 16;
        const _Float16* b1p = b0p + (size_t)16 * D_DIM;

        v8f c0 = {}, c1 = {};
        v16h a  = LOAD_A(0);
        v16h b0 = *(const v16h*)(b0p);
        v16h b1 = *(const v16h*)(b1p);

#pragma unroll 4
        for (int k0 = 32; k0 < D_DIM; k0 += 32) {
            v16h an  = LOAD_A(k0);
            v16h b0n = *(const v16h*)(b0p + k0);
            v16h b1n = *(const v16h*)(b1p + k0);
            // Pull next dual-tile's B lines toward the caches (speculative, no guard).
            __builtin_prefetch((const void*)(b0p + (size_t)256 * D_DIM + k0), 0, 3);
            __builtin_prefetch((const void*)(b1p + (size_t)256 * D_DIM + k0), 0, 3);
            c0 = __builtin_amdgcn_wmma_f32_16x16x32_f16(
                     false, a, false, b0, (short)0, c0, false, false);
            c1 = __builtin_amdgcn_wmma_f32_16x16x32_f16(
                     false, a, false, b1, (short)0, c1, false, false);
            a = an; b0 = b0n; b1 = b1n;
        }
        c0 = __builtin_amdgcn_wmma_f32_16x16x32_f16(
                 false, a, false, b0, (short)0, c0, false, false);
        c1 = __builtin_amdgcn_wmma_f32_16x16x32_f16(
                 false, a, false, b1, (short)0, c1, false, false);

        // C layout: cX[r] = cos( row mbase + r + 8*hi , col nX )
        const int n0 = cb + m;
        const int n1 = cb + 16 + m;
#pragma unroll
        for (int r = 0; r < 8; ++r) {
            float vm; int nm;
            if (c1[r] > c0[r]) { vm = c1[r]; nm = n1; }
            else               { vm = c0[r]; nm = n0; }
            if (vm > best[r]) { best[r] = vm; bidx[r] = nm; }
        }
    }
#undef LOAD_A

    // Cross-lane argmax within each 16-lane half (rows split low/high halves).
#pragma unroll
    for (int off = 8; off > 0; off >>= 1) {
#pragma unroll
        for (int r = 0; r < 8; ++r) {
            float ov = __shfl_xor(best[r], off);
            int   oi = __shfl_xor(bidx[r], off);
            if (ov > best[r] || (ov == best[r] && oi < bidx[r])) {
                best[r] = ov; bidx[r] = oi;
            }
        }
    }
    if (lane == 0) {
#pragma unroll
        for (int r = 0; r < 8; ++r) { red_val[wv][r] = best[r]; red_idx[wv][r] = bidx[r]; }
    }
    if (lane == 16) {
#pragma unroll
        for (int r = 0; r < 8; ++r) { red_val[wv][8 + r] = best[r]; red_idx[wv][8 + r] = bidx[r]; }
    }
    __syncthreads();

    // Final cross-wave reduce: thread t handles row mbase + t.
    if (tid < 16) {
        float bv = -3.4e38f;
        int   bi = 0x7FFFFFFF;
#pragma unroll
        for (int w = 0; w < 8; ++w) {
            float v = red_val[w][tid];
            int   i = red_idx[w][tid];
            if (v > bv || (v == bv && i < bi)) { bv = v; bi = i; }
        }
        kidx[mbase + tid] = bi;
    }
}

// ---------------------------------------------------------------------------
// Stage 3: gather e_p[k_idx] -> Ek | Ev, plus x_block passthrough.
// One block per query row; b128 streaming copies.
// ---------------------------------------------------------------------------
__global__ __launch_bounds__(256)
void gather_out(const float* __restrict__ ep,
                const float* __restrict__ xb,
                const int* __restrict__ kidx,
                float* __restrict__ out, int B) {
    const int b   = blockIdx.x;
    const int tid = threadIdx.x;
    const int idx = kidx[b];

    const float4* src = (const float4*)(ep + (size_t)idx * 8 * D_DIM);
    float4* outEk = (float4*)out + (size_t)b * 4 * (D_DIM / 4);
    float4* outEv = (float4*)out + (size_t)B * 4 * (D_DIM / 4) + (size_t)b * 4 * (D_DIM / 4);

    // 8 rows * 192 float4 = 1536 vectors; first 768 -> Ek, rest -> Ev.
    for (int t = tid; t < 8 * (D_DIM / 4); t += 256) {
        float4 v = src[t];
        if (t < 4 * (D_DIM / 4)) outEk[t] = v;
        else                     outEv[t - 4 * (D_DIM / 4)] = v;
    }

    const float4* xsrc = (const float4*)xb + (size_t)b * (D_DIM / 4);
    float4* outX = (float4*)out + (size_t)2 * B * 4 * (D_DIM / 4) + (size_t)b * (D_DIM / 4);
    if (tid < (D_DIM / 4)) outX[tid] = xsrc[tid];
}

// ---------------------------------------------------------------------------
extern "C" void kernel_launch(void* const* d_in, const int* in_sizes, int n_in,
                              void* d_out, int out_size, void* d_ws, size_t ws_size,
                              hipStream_t stream) {
    (void)n_in; (void)out_size; (void)ws_size;
    const float* xq = (const float*)d_in[0];   // (B, 768)
    const float* xb = (const float*)d_in[1];   // (B, 768)
    const float* ek = (const float*)d_in[2];   // (P, 768)
    const float* ep = (const float*)d_in[3];   // (P, 8, 768)

    const int B = in_sizes[0] / D_DIM;         // 8192
    const int P = in_sizes[2] / D_DIM;         // 4096

    // Workspace layout: q_f16 | nK_f16 | k_idx
    _Float16* qh  = (_Float16*)d_ws;
    _Float16* nkh = qh + (size_t)B * D_DIM;
    int*      kid = (int*)(nkh + (size_t)P * D_DIM);

    rownorm_f16<<<(B + 7) / 8, 256, 0, stream>>>(xq, qh, B);
    rownorm_f16<<<(P + 7) / 8, 256, 0, stream>>>(ek, nkh, P);
    argmax_gemm<<<B / 16, 256, 0, stream>>>(qh, nkh, kid, P);
    gather_out<<<B, 256, 0, stream>>>(ep, xb, kid, (float*)d_out, B);
}
